// MoELayer_2250562863258
// MI455X (gfx1250) — compile-verified, matrix operand order checked by reading, not run on hardware
//
#include <hip/hip_runtime.h>
#include <math.h>

#define N_TOK 8192
#define DIM   1024
#define EXP   8
#define HID   4096

typedef __attribute__((ext_vector_type(16))) __bf16 v16bf;
typedef __attribute__((ext_vector_type(8)))  float  v8f;

#if defined(__has_builtin)
#if __has_builtin(__builtin_amdgcn_tensor_load_to_lds) && \
    __has_builtin(__builtin_amdgcn_s_wait_tensorcnt)
#define USE_TDM 1
#endif
#endif

#ifdef USE_TDM
typedef __attribute__((ext_vector_type(4))) unsigned int v4u;
typedef __attribute__((ext_vector_type(8))) int v8i;
typedef __attribute__((ext_vector_type(4))) int v4i;

// TDM load of a [128 rows x 32 cols] bf16 tile from a row-major [*, HID]
// tensor into LDS at byte offset lds_off. ga = global byte address of tile
// start. D# layout per CDNA5 ISA ch. 8 (group0 + group1; groups 2/3 zero).
__device__ __forceinline__ void tdm_load_tile128x32(unsigned lds_off,
                                                    unsigned long long ga,
                                                    int row_pitch_elems,
                                                    int n_rows_tensor) {
  v4u g0;
  g0[0] = 1u;                                    // count=1 (valid descriptor)
  g0[1] = lds_off;                               // lds_addr
  g0[2] = (unsigned)ga;                          // global_addr[31:0]
  g0[3] = (unsigned)((ga >> 32) & 0x01ffffffu)   // global_addr[56:32]
          | 0x80000000u;                         // type=2 ("image")
  v8i g1;
  g1[0] = 1 << 16;                               // data_size=1 (2 bytes)
  g1[1] = (int)((unsigned)row_pitch_elems << 16);        // tensor_dim0 lo16
  g1[2] = (int)(((unsigned)row_pitch_elems >> 16) |
                ((unsigned)n_rows_tensor << 16));        // dim0 hi / dim1 lo
  g1[3] = (int)(((unsigned)n_rows_tensor >> 16) |
                (32u << 16));                            // dim1 hi / tile_dim0=32
  g1[4] = 128;                                   // tile_dim1=128, tile_dim2=0
  g1[5] = row_pitch_elems;                       // tensor_dim0_stride lo32
  g1[6] = 0;
  g1[7] = 0;
  v4i gz4 = {0, 0, 0, 0};
  v8i gz8 = {0, 0, 0, 0, 0, 0, 0, 0};
  __builtin_amdgcn_tensor_load_to_lds(g0, g1, gz4, gz4, gz8, 0);
}
#endif

__device__ __forceinline__ float gelu_exact(float x) {
  return 0.5f * x * (1.0f + erff(x * 0.70710678118654752f));
}

// ---------------------------------------------------------------------------
// Gating: logits = x @ gate_w + gate_b ; top-2 softmax -> combine[N, E]
// ---------------------------------------------------------------------------
__global__ __launch_bounds__(256) void gate_kernel(
    const float* __restrict__ x, const float* __restrict__ gw,
    const float* __restrict__ gb, float* __restrict__ combine) {
  __shared__ float gws[DIM * EXP];
  int tid = threadIdx.x;
  for (int i = tid; i < DIM * EXP; i += 256) gws[i] = gw[i];
  __syncthreads();

  int token = blockIdx.x * 256 + tid;
  float acc[EXP];
#pragma unroll
  for (int e = 0; e < EXP; ++e) acc[e] = gb[e];
  const float* xr = x + (size_t)token * DIM;
  for (int d = 0; d < DIM; d += 4) {
    float4 xv = *(const float4*)(xr + d);
#pragma unroll
    for (int e = 0; e < EXP; ++e) {
      acc[e] += xv.x * gws[(d + 0) * EXP + e];
      acc[e] += xv.y * gws[(d + 1) * EXP + e];
      acc[e] += xv.z * gws[(d + 2) * EXP + e];
      acc[e] += xv.w * gws[(d + 3) * EXP + e];
    }
  }
  int i1 = 0; float m1 = acc[0];
#pragma unroll
  for (int e = 1; e < EXP; ++e) if (acc[e] > m1) { m1 = acc[e]; i1 = e; }
  int i2 = -1; float m2 = -INFINITY;
#pragma unroll
  for (int e = 0; e < EXP; ++e)
    if (e != i1 && acc[e] > m2) { m2 = acc[e]; i2 = e; }
  float e2 = __expf(m2 - m1);
  float p1 = 1.0f / (1.0f + e2);
  float p2 = e2 * p1;
  float* c = combine + (size_t)token * EXP;
#pragma unroll
  for (int e = 0; e < EXP; ++e) c[e] = 0.0f;
  c[i1] = p1;
  c[i2] = p2;
}

__global__ void zero_counts(int* counts) {
  if (threadIdx.x < EXP) counts[threadIdx.x] = 0;
}

__global__ __launch_bounds__(256) void build_lists(
    const float* __restrict__ combine, int* __restrict__ counts,
    int* __restrict__ idx) {
  int token = blockIdx.x * 256 + threadIdx.x;
#pragma unroll
  for (int e = 0; e < EXP; ++e) {
    if (combine[(size_t)token * EXP + e] > 0.0f) {
      int slot = atomicAdd(&counts[e], 1);
      idx[(size_t)e * N_TOK + slot] = token;
    }
  }
}

// ---------------------------------------------------------------------------
// GEMM1: H[slot, HID] = gelu( gather(x)[slot] @ W1 + b1 ), bf16 out
// 128x128 tile, K-step 32, double-buffered LDS, 1 barrier per K-step.
// ---------------------------------------------------------------------------
__global__ __launch_bounds__(256) void gemm1_kernel(
    const float* __restrict__ X, const float* __restrict__ W,
    const float* __restrict__ bias, __bf16* __restrict__ H,
    const int* __restrict__ idx, const int* __restrict__ cnt_ptr,
    int cnt_const) {
  int count = cnt_ptr ? *cnt_ptr : cnt_const;
  int mbase = blockIdx.y * 128;
  if (mbase >= count) return;
  int nbase = blockIdx.x * 128;

  __shared__ __bf16 As[2][128 * 32];  // [m][k]
  __shared__ __bf16 Bs[2][128 * 32];  // [n][k]
  __shared__ int rowTok[128];

  int tid = threadIdx.x;
  for (int r = tid; r < 128; r += 256) {
    int slot = mbase + r;
    rowTok[r] = (slot < count) ? (idx ? idx[slot] : slot) : 0;
  }
  __syncthreads();

  int lane = tid & 31;
  int wid = tid >> 5;
  int wm = wid >> 2;      // 0..1
  int wn = wid & 3;       // 0..3
  int l16 = lane & 15;
  int khalf = lane >> 4;

  v8f acc[4][2];
#pragma unroll
  for (int i = 0; i < 4; ++i)
#pragma unroll
    for (int j = 0; j < 2; ++j)
#pragma unroll
      for (int q = 0; q < 8; ++q) acc[i][j][q] = 0.0f;

  int ar = tid >> 1;
  int ah = (tid & 1) * 16;
  int bk = tid >> 3;
  int bn = (tid & 7) * 16;

  auto stageA = [&](int kb, int buf) {
    const float* src = X + (size_t)rowTok[ar] * DIM + kb + ah;
    if (kb + 32 < DIM) __builtin_prefetch(src + 32, 0, 1);
    __bf16* dst = &As[buf][ar * 32 + ah];
#pragma unroll
    for (int j = 0; j < 16; j += 4) {
      float4 v = *(const float4*)(src + j);
      dst[j + 0] = (__bf16)v.x; dst[j + 1] = (__bf16)v.y;
      dst[j + 2] = (__bf16)v.z; dst[j + 3] = (__bf16)v.w;
    }
  };
  auto stageB = [&](int kb, int buf) {
    const float* src = W + (size_t)(kb + bk) * HID + nbase + bn;
    if (kb + 32 < DIM) __builtin_prefetch(src + (size_t)32 * HID, 0, 1);
#pragma unroll
    for (int j = 0; j < 16; j += 4) {
      float4 v = *(const float4*)(src + j);
      Bs[buf][(bn + j + 0) * 32 + bk] = (__bf16)v.x;
      Bs[buf][(bn + j + 1) * 32 + bk] = (__bf16)v.y;
      Bs[buf][(bn + j + 2) * 32 + bk] = (__bf16)v.z;
      Bs[buf][(bn + j + 3) * 32 + bk] = (__bf16)v.w;
    }
  };

  stageA(0, 0);
  stageB(0, 0);
  __syncthreads();

  for (int kb = 0; kb < DIM; kb += 32) {
    int cur = (kb >> 5) & 1;
    if (kb + 32 < DIM) {  // stage next tile into other buffer
      stageA(kb + 32, cur ^ 1);
      stageB(kb + 32, cur ^ 1);
    }
    // hoist all fragments so the 12 ds_loads issue before the WMMA chain
    v16bf bfr[2], afr[4];
#pragma unroll
    for (int fn = 0; fn < 2; ++fn)
      bfr[fn] =
          *(const v16bf*)&Bs[cur][(wn * 32 + fn * 16 + l16) * 32 + khalf * 16];
#pragma unroll
    for (int fm = 0; fm < 4; ++fm)
      afr[fm] =
          *(const v16bf*)&As[cur][(wm * 64 + fm * 16 + l16) * 32 + khalf * 16];
#pragma unroll
    for (int fm = 0; fm < 4; ++fm)
#pragma unroll
      for (int fn = 0; fn < 2; ++fn)
        acc[fm][fn] = __builtin_amdgcn_wmma_f32_16x16x32_bf16(
            false, afr[fm], false, bfr[fn], (short)0, acc[fm][fn], false,
            false);
    __syncthreads();
  }

#pragma unroll
  for (int fm = 0; fm < 4; ++fm) {
    int slot0 = mbase + wm * 64 + fm * 16 + khalf * 8;
#pragma unroll
    for (int fn = 0; fn < 2; ++fn) {
      int ncol = nbase + wn * 32 + fn * 16 + l16;
      float b = bias[ncol];
#pragma unroll
      for (int vv = 0; vv < 8; ++vv) {
        int srow = slot0 + vv;
        if (srow < count)
          H[(size_t)srow * HID + ncol] = (__bf16)gelu_exact(acc[fm][fn][vv] + b);
      }
    }
  }
}

// ---------------------------------------------------------------------------
// GEMM2: out[tok] (+)= combine[tok,e] * ( H[slot] @ W2 + b2 )
// A-tiles (bf16, contiguous) staged by the Tensor Data Mover when available;
// DMA of tile i+1 overlaps WMMA on tile i (double-buffered, s_wait_tensorcnt).
// ---------------------------------------------------------------------------
__global__ __launch_bounds__(256) void gemm2_kernel(
    const __bf16* __restrict__ H, const float* __restrict__ W,
    const float* __restrict__ bias, const float* __restrict__ combine,
    int expert_id, float* __restrict__ out, const int* __restrict__ idx,
    const int* __restrict__ cnt_ptr, int cnt_const, int accumulate) {
  int count = cnt_ptr ? *cnt_ptr : cnt_const;
  int mbase = blockIdx.y * 128;
  if (mbase >= count) return;
  int nbase = blockIdx.x * 128;

  __shared__ __bf16 As[2][128 * 32];
  __shared__ __bf16 Bs[2][128 * 32];
  __shared__ int rowTok[128];

  int tid = threadIdx.x;
  for (int r = tid; r < 128; r += 256) {
    int slot = mbase + r;
    rowTok[r] = (slot < count) ? (idx ? idx[slot] : slot) : 0;
  }
  __syncthreads();

  int lane = tid & 31;
  int wid = tid >> 5;
  int wm = wid >> 2;
  int wn = wid & 3;
  int l16 = lane & 15;
  int khalf = lane >> 4;

  v8f acc[4][2];
#pragma unroll
  for (int i = 0; i < 4; ++i)
#pragma unroll
    for (int j = 0; j < 2; ++j)
#pragma unroll
      for (int q = 0; q < 8; ++q) acc[i][j][q] = 0.0f;

  int ar = tid >> 1;
  int ah = (tid & 1) * 16;
  int bk = tid >> 3;
  int bn = (tid & 7) * 16;

#ifdef USE_TDM
  unsigned ldsA[2] = {(unsigned)(size_t)&As[0][0], (unsigned)(size_t)&As[1][0]};
  unsigned long long gH =
      (unsigned long long)(size_t)H + (size_t)mbase * HID * 2ull;
#else
  auto stageA = [&](int kb, int buf) {
    const __bf16* src = H + (size_t)(mbase + ar) * HID + kb + ah;
    uint4* dst = (uint4*)&As[buf][ar * 32 + ah];
    dst[0] = ((const uint4*)src)[0];
    dst[1] = ((const uint4*)src)[1];
  };
#endif
  auto stageB = [&](int kb, int buf) {
    const float* src = W + (size_t)(kb + bk) * DIM + nbase + bn;
    if (kb + 32 < HID) __builtin_prefetch(src + (size_t)32 * DIM, 0, 1);
#pragma unroll
    for (int j = 0; j < 16; j += 4) {
      float4 v = *(const float4*)(src + j);
      Bs[buf][(bn + j + 0) * 32 + bk] = (__bf16)v.x;
      Bs[buf][(bn + j + 1) * 32 + bk] = (__bf16)v.y;
      Bs[buf][(bn + j + 2) * 32 + bk] = (__bf16)v.z;
      Bs[buf][(bn + j + 3) * 32 + bk] = (__bf16)v.w;
    }
  };

  // prologue: stage tile 0
#ifdef USE_TDM
  if (tid < 32) tdm_load_tile128x32(ldsA[0], gH, HID, N_TOK);
#else
  stageA(0, 0);
#endif
  stageB(0, 0);
#ifdef USE_TDM
  if (tid < 32) __builtin_amdgcn_s_wait_tensorcnt(0);
#endif
  __syncthreads();

  for (int kb = 0; kb < HID; kb += 32) {
    int cur = (kb >> 5) & 1;
    bool have_next = (kb + 32) < HID;
    if (have_next) {
#ifdef USE_TDM
      if (tid < 32)
        tdm_load_tile128x32(ldsA[cur ^ 1], gH + (size_t)(kb + 32) * 2ull, HID,
                            N_TOK);
#else
      stageA(kb + 32, cur ^ 1);
#endif
      stageB(kb + 32, cur ^ 1);
    }
    v16bf bfr[2], afr[4];
#pragma unroll
    for (int fn = 0; fn < 2; ++fn)
      bfr[fn] =
          *(const v16bf*)&Bs[cur][(wn * 32 + fn * 16 + l16) * 32 + khalf * 16];
#pragma unroll
    for (int fm = 0; fm < 4; ++fm)
      afr[fm] =
          *(const v16bf*)&As[cur][(wm * 64 + fm * 16 + l16) * 32 + khalf * 16];
#pragma unroll
    for (int fm = 0; fm < 4; ++fm)
#pragma unroll
      for (int fn = 0; fn < 2; ++fn)
        acc[fm][fn] = __builtin_amdgcn_wmma_f32_16x16x32_bf16(
            false, afr[fm], false, bfr[fn], (short)0, acc[fm][fn], false,
            false);
#ifdef USE_TDM
    if (have_next && tid < 32) __builtin_amdgcn_s_wait_tensorcnt(0);
#endif
    __syncthreads();
  }

#pragma unroll
  for (int fm = 0; fm < 4; ++fm) {
    int r0 = wm * 64 + fm * 16 + khalf * 8;
#pragma unroll
    for (int fn = 0; fn < 2; ++fn) {
      int ncol = nbase + wn * 32 + fn * 16 + l16;
      float b = bias[ncol];
#pragma unroll
      for (int vv = 0; vv < 8; ++vv) {
        int srow = mbase + r0 + vv;
        if (srow < count) {
          int tok = rowTok[r0 + vv];
          float val = acc[fm][fn][vv] + b;
          size_t o = (size_t)tok * DIM + ncol;
          if (accumulate) {
            float g = combine[(size_t)tok * EXP + expert_id];
            out[o] += g * val;
          } else {
            out[o] = val;
          }
        }
      }
    }
  }
}

// ---------------------------------------------------------------------------
extern "C" void kernel_launch(void* const* d_in, const int* in_sizes, int n_in,
                              void* d_out, int out_size, void* d_ws,
                              size_t ws_size, hipStream_t stream) {
  const float* x      = (const float*)d_in[0];
  const float* gate_w = (const float*)d_in[1];
  const float* gate_b = (const float*)d_in[2];
  const float* w1     = (const float*)d_in[3];
  const float* b1     = (const float*)d_in[4];
  const float* w2     = (const float*)d_in[5];
  const float* b2     = (const float*)d_in[6];
  const float* ws1    = (const float*)d_in[7];
  const float* bs1    = (const float*)d_in[8];
  const float* ws2    = (const float*)d_in[9];
  const float* bs2    = (const float*)d_in[10];
  float* out = (float*)d_out;

  char* p = (char*)d_ws;
  float* combine = (float*)p;  p += (size_t)N_TOK * EXP * sizeof(float);
  int*   counts  = (int*)p;    p += 256;
  int*   idx     = (int*)p;    p += (size_t)EXP * N_TOK * sizeof(int);
  __bf16* H      = (__bf16*)p;  // N_TOK * HID * 2 bytes

  gate_kernel<<<N_TOK / 256, 256, 0, stream>>>(x, gate_w, gate_b, combine);
  zero_counts<<<1, 32, 0, stream>>>(counts);
  build_lists<<<N_TOK / 256, 256, 0, stream>>>(combine, counts, idx);

  dim3 blk(256);
  dim3 g1(HID / 128, N_TOK / 128);  // 32 x 64
  dim3 g2(DIM / 128, N_TOK / 128);  //  8 x 64

  // shared expert: dense over all tokens, writes out
  gemm1_kernel<<<g1, blk, 0, stream>>>(x, ws1, bs1, H, nullptr, nullptr, N_TOK);
  gemm2_kernel<<<g2, blk, 0, stream>>>(H, ws2, bs2, nullptr, 0, out, nullptr,
                                       nullptr, N_TOK, 0);

  // routed experts: gathered tokens, accumulate into out
  for (int e = 0; e < EXP; ++e) {
    gemm1_kernel<<<g1, blk, 0, stream>>>(
        x, w1 + (size_t)e * DIM * HID, b1 + (size_t)e * HID, H,
        idx + (size_t)e * N_TOK, counts + e, 0);
    gemm2_kernel<<<g2, blk, 0, stream>>>(
        H, w2 + (size_t)e * HID * DIM, b2 + (size_t)e * DIM, combine, e, out,
        idx + (size_t)e * N_TOK, counts + e, 0, 1);
  }
}
